// MultiHeadAttention_25391846654154
// MI455X (gfx1250) — matrix operand order repaired and checked by reading, not compiled, not run on hardware
//
#include <hip/hip_runtime.h>
#include <hip/hip_bf16.h>
#include <cstdint>
#include <cstddef>

// ---------------- problem constants (from reference) ----------------
constexpr int kB    = 2;
constexpr int kS    = 2048;
constexpr int kD    = 2048;     // D_MODEL
constexpr int kH    = 16;       // N_HEADS
constexpr int kKVH  = 4;        // N_KV_HEADS
constexpr int kDk   = 128;      // D_K
constexpr int kWin  = 64;       // WINDOW/2
constexpr int kRows = kB * kS;  // GEMM M = 4096
constexpr int kKVD  = kKVH * kDk; // 512

typedef __bf16 bf16;
typedef __attribute__((ext_vector_type(4)))  __bf16 v4bf;
typedef __attribute__((ext_vector_type(8)))  __bf16 v8bf;
typedef __attribute__((ext_vector_type(16))) __bf16 v16bf;
typedef __attribute__((ext_vector_type(8)))  float  v8f;
typedef __attribute__((ext_vector_type(4)))  unsigned int v4u;
typedef __attribute__((ext_vector_type(8)))  int v8i;
typedef __attribute__((ext_vector_type(4)))  int v4i;

#ifndef __has_builtin
#define __has_builtin(x) 0
#endif
#if __has_builtin(__builtin_amdgcn_tensor_load_to_lds) && \
    __has_builtin(__builtin_amdgcn_s_wait_tensorcnt)
#define USE_TDM 1
#else
#define USE_TDM 0
#endif

__device__ __forceinline__ v16bf cat8(v8bf lo, v8bf hi) {
  return __builtin_shufflevector(lo, hi, 0,1,2,3,4,5,6,7,8,9,10,11,12,13,14,15);
}

#if USE_TDM
// 2-D Tensor-DMA load of a (tile_d1 rows x tile_d0 elems) bf16 tile into LDS.
// D# built per CDNA5 ISA; groups 2/3 zero (2-D tensor).
// LDS byte address: flat->LDS mapping keeps addr[31:0] (ISA 10.2), so the low 32 bits
// of the generic shared pointer are the LDS offset.
__device__ __forceinline__ void tdm_load_tile_2d(void* lds_ptr, const void* gptr,
                                                 unsigned tile_d0, unsigned tile_d1,
                                                 unsigned long long stride0_elems) {
  const unsigned long long ga = (unsigned long long)(size_t)gptr;
  v4u g0;
  g0.x = 1u;                                              // count=1, user-mode, no gather
  g0.y = (unsigned)(size_t)lds_ptr;                       // lds_addr (bytes)
  g0.z = (unsigned)ga;                                    // global_addr[31:0]
  g0.w = (unsigned)((ga >> 32) & 0x01FFFFFFull) | (2u << 30);  // global_addr[56:32], type=2
  v8i g1;
  g1[0] = (int)(1u << 16);                                // mask=0, data_size=1 (2 bytes)
  g1[1] = (int)((tile_d0 & 0xFFFFu) << 16);               // tensor_dim0 = tile_d0
  g1[2] = (int)((tile_d0 >> 16) | ((tile_d1 & 0xFFFFu) << 16)); // tensor_dim1 = tile_d1
  g1[3] = (int)((tile_d1 >> 16) | (tile_d0 << 16));       // tile_dim0
  g1[4] = (int)(tile_d1 & 0xFFFFu);                       // tile_dim1 (tile_dim2=0)
  g1[5] = (int)(stride0_elems & 0xFFFFFFFFull);           // tensor_dim0_stride lo
  g1[6] = (int)((stride0_elems >> 32) & 0xFFFFull);       // stride hi (dim1 stride unused)
  g1[7] = 0;
  const v4i z4 = {0, 0, 0, 0};
#if defined(__clang_major__) && (__clang_major__ >= 23)
  const v8i z8 = {0, 0, 0, 0, 0, 0, 0, 0};
  __builtin_amdgcn_tensor_load_to_lds(g0, g1, z4, z4, z8, 0);
#else
  __builtin_amdgcn_tensor_load_to_lds(g0, g1, z4, z4, 0);
#endif
}
#endif

// ---------------- f32 -> bf16 bulk convert ----------------
__global__ __launch_bounds__(256) void cvt_f32_to_bf16(const float* __restrict__ in,
                                                       bf16* __restrict__ out, int n4) {
  int i = blockIdx.x * blockDim.x + threadIdx.x;
  if (i >= n4) return;
  float4 v = ((const float4*)in)[i];
  v4bf o;
  o.x = (bf16)v.x; o.y = (bf16)v.y; o.z = (bf16)v.z; o.w = (bf16)v.w;
  ((v4bf*)out)[i] = o;
}

// ---------------- bf16 WMMA GEMM:  C[M,N] = A[M,K] * W[N,K]^T + bias ----------------
// Workgroup tile 128x128 (8 waves, 4M x 2N; wave tile 32x64 = 2x4 WMMA accumulators).
// K-slabs of 32 are staged into double-buffered LDS by the Tensor Data Mover (wave 0
// issues two 2-D TDM descriptors per slab and gates the block with s_wait_tensorcnt),
// then operand fragments are fetched with ds_load_b128 per the CDNA5 16-bit A(16x32) /
// B(32x16) lane layouts (ISA 7.12.2).
__global__ __launch_bounds__(256) void gemm_bf16_wmma(const bf16* __restrict__ A,
                                                      const bf16* __restrict__ W,
                                                      const float* __restrict__ bias,
                                                      void* __restrict__ C,
                                                      int M, int N, int K, int outBf16) {
  __shared__ bf16 sA[2][128 * 32];
  __shared__ bf16 sB[2][128 * 32];

  const int lane  = threadIdx.x & 31;
  const int wave  = threadIdx.x >> 5;
  const int waveM = wave & 3;
  const int waveN = wave >> 2;
  const int mBase = blockIdx.x * 128 + waveM * 32;
  const int nBase = blockIdx.y * 128 + waveN * 64;
  const int sel   = lane >> 4;    // half-wave select
  const int lid   = lane & 15;

  const bf16* Ab = A + (size_t)blockIdx.x * 128 * K;
  const bf16* Bb = W + (size_t)blockIdx.y * 128 * K;

  v8f acc[2][4];
  const v8f vzero = {0.f,0.f,0.f,0.f,0.f,0.f,0.f,0.f};
#pragma unroll
  for (int mi = 0; mi < 2; ++mi)
#pragma unroll
    for (int ni = 0; ni < 4; ++ni) acc[mi][ni] = vzero;

#if !USE_TDM
  auto stage = [&](int b2, int kk) {
    for (int i = threadIdx.x; i < (128 * 32) / 8; i += 256) {
      const int row = i >> 2, ch = (i & 3) * 8;
      *(v8bf*)&sA[b2][row * 32 + ch] = *(const v8bf*)(Ab + (size_t)row * K + kk + ch);
      *(v8bf*)&sB[b2][row * 32 + ch] = *(const v8bf*)(Bb + (size_t)row * K + kk + ch);
    }
  };
#endif

  // prologue: stage slab 0 into buffer 0
#if USE_TDM
  if (wave == 0) {
    tdm_load_tile_2d(&sA[0][0], Ab, 32, 128, (unsigned long long)K);
    tdm_load_tile_2d(&sB[0][0], Bb, 32, 128, (unsigned long long)K);
    __builtin_amdgcn_s_wait_tensorcnt(0);
  }
#else
  stage(0, 0);
#endif
  __syncthreads();

  int buf = 0;
  for (int k = 0; k < K; k += 32) {
    const int kn = k + 32;
#if USE_TDM
    if (kn < K && wave == 0) {
      tdm_load_tile_2d(&sA[buf ^ 1][0], Ab + kn, 32, 128, (unsigned long long)K);
      tdm_load_tile_2d(&sB[buf ^ 1][0], Bb + kn, 32, 128, (unsigned long long)K);
    }
#else
    if (kn < K) stage(buf ^ 1, kn);
#endif

    const bf16* sAb = &sA[buf][0];
    const bf16* sBb = &sB[buf][0];
    v16bf a[2], b[4];
#pragma unroll
    for (int mi = 0; mi < 2; ++mi) {
      // A 16x32 bf16 layout: lane<16 -> M=lid, K {0..7,16..23}; lane>=16 -> K {8..15,24..31}
      const int r = (waveM * 32 + mi * 16 + lid) * 32;
      a[mi] = cat8(*(const v8bf*)&sAb[r + sel * 8],
                   *(const v8bf*)&sAb[r + 16 + sel * 8]);
    }
#pragma unroll
    for (int ni = 0; ni < 4; ++ni) {
      // B 32x16 bf16 layout: lane<16 -> N=lid, K {0..7,16..23}; lane>=16 -> K {8..15,24..31}
      const int r = (waveN * 64 + ni * 16 + lid) * 32 + sel * 8;
      b[ni] = cat8(*(const v8bf*)&sBb[r], *(const v8bf*)&sBb[r + 16]);
    }
#pragma unroll
    for (int mi = 0; mi < 2; ++mi)
#pragma unroll
      for (int ni = 0; ni < 4; ++ni)
        acc[mi][ni] = __builtin_amdgcn_wmma_f32_16x16x32_bf16(
            false, a[mi], false, b[ni], (short)0, acc[mi][ni], false, false);

#if USE_TDM
    if (wave == 0) __builtin_amdgcn_s_wait_tensorcnt(0);
#endif
    __syncthreads();
    buf ^= 1;
  }

  // Epilogue. C 16x16 f32 layout: VGPR r -> M = r (lanes 0-15) / r+8 (lanes 16-31); N = lid.
#pragma unroll
  for (int ni = 0; ni < 4; ++ni) {
    const int col  = nBase + ni * 16 + lid;
    const float bc = bias[col];
#pragma unroll
    for (int mi = 0; mi < 2; ++mi) {
#pragma unroll
      for (int r = 0; r < 8; ++r) {
        const int row = mBase + mi * 16 + r + sel * 8;
        const float val = acc[mi][ni][r] + bc;
        if (outBf16) ((bf16*)C)[(size_t)row * N + col] = (bf16)val;
        else         ((float*)C)[(size_t)row * N + col] = val;
      }
    }
  }
}

// ---------------- fused RMSNorm + RoPE, f32 in -> bf16 out ----------------
// One wave32 per (b,s,head) vector of 128 elems (4 per lane).
__global__ __launch_bounds__(256) void normrope_kernel(const float* __restrict__ xin,
                                                       const float* __restrict__ w,
                                                       bf16* __restrict__ xout,
                                                       int H, int total) {
  const int vec  = (blockIdx.x * blockDim.x + threadIdx.x) >> 5;
  const int lane = threadIdx.x & 31;
  if (vec >= total) return;
  const int spos = (vec / H) % kS;

  const float4 xv = *(const float4*)(xin + (size_t)vec * kDk + lane * 4);
  float ss = xv.x*xv.x + xv.y*xv.y + xv.z*xv.z + xv.w*xv.w;
#pragma unroll
  for (int o = 16; o; o >>= 1) ss += __shfl_xor(ss, o);
  const float rinv = 1.0f / sqrtf(ss * (1.0f / (float)kDk) + 1e-8f);

  const float4 wv = *(const float4*)(w + lane * 4);
  float x0 = xv.x * rinv * wv.x;
  float x1 = xv.y * rinv * wv.y;
  float x2 = xv.z * rinv * wv.z;
  float x3 = xv.w * rinv * wv.w;

  // pair A uses freq index iA=2*lane, pair B uses iB=2*lane+1 (inv_freq = base^(-2i/dk))
  const int iA = 2 * lane, iB = 2 * lane + 1;
  float fA = powf(10000.0f, -(float)(2 * iA) / (float)kDk);
  float fB = powf(10000.0f, -(float)(2 * iB) / (float)kDk);
  float sA, cA, sB, cB;
  sincosf((float)spos * fA, &sA, &cA);
  sincosf((float)spos * fB, &sB, &cB);

  v4bf o;
  o.x = (bf16)(x0 * cA - x1 * sA);
  o.y = (bf16)(x0 * sA + x1 * cA);
  o.z = (bf16)(x2 * cB - x3 * sB);
  o.w = (bf16)(x2 * sB + x3 * cB);
  *(v4bf*)(xout + (size_t)vec * kDk + lane * 4) = o;
}

// ---------------- windowed causal attention ----------------
// One block per (b, head, 16-query tile). Keys [q0-64, q0+15] (80 rows) staged in LDS.
// Warp w handles queries 2w and 2w+1. Entries the reference fills with -1e4 underflow to
// exactly 0 after exp in f32, so restricting to the allowed window matches bit-for-bit.
__global__ __launch_bounds__(256) void attn_kernel(const bf16* __restrict__ qb,
                                                   const bf16* __restrict__ kb,
                                                   const bf16* __restrict__ vb,
                                                   const float* __restrict__ scale_p,
                                                   bf16* __restrict__ ob) {
  const int bid = blockIdx.x;
  const int qt  = bid & (kS / 16 - 1);
  const int h   = (bid >> 7) & (kH - 1);
  const int b   = bid >> 11;
  const int q0  = qt * 16;
  const int kstart = q0 - kWin;          // may be negative
  const int kvh = h >> 2;                // N_REP = 4

  __shared__ float sQ[16][kDk];
  __shared__ bf16  sK[80][kDk];
  __shared__ bf16  sV[80][kDk];
  __shared__ float sP[16][80];

  const float scale = *scale_p;

  for (int i = threadIdx.x; i < 16 * kDk; i += 256) {
    const int qi = i >> 7, d = i & 127;
    sQ[qi][d] = (float)qb[(((size_t)(b * kS + q0 + qi)) * kH + h) * kDk + d];
  }
  for (int i = threadIdx.x; i < 80 * 32; i += 256) {
    const int lk = i >> 5;
    const int d4 = (i & 31) * 4;
    const int j  = kstart + lk;
    v4bf kvv, vvv;
    if (j >= 0) {
      kvv = *(const v4bf*)(kb + (((size_t)(b * kS + j)) * kKVH + kvh) * kDk + d4);
      vvv = *(const v4bf*)(vb + (((size_t)(b * kS + j)) * kKVH + kvh) * kDk + d4);
    } else {
      kvv.x = (bf16)0.f; kvv.y = (bf16)0.f; kvv.z = (bf16)0.f; kvv.w = (bf16)0.f;
      vvv = kvv;
    }
    *(v4bf*)&sK[lk][d4] = kvv;
    *(v4bf*)&sV[lk][d4] = vvv;
  }
  __syncthreads();

  const int wave = threadIdx.x >> 5;
  const int lane = threadIdx.x & 31;

  for (int qq = 0; qq < 2; ++qq) {
    const int qi    = wave * 2 + qq;
    const int qglob = q0 + qi;
    const int negks = -kstart;
    const int lkmin = qi > negks ? qi : negks;   // j>=0 and j>=q-64
    const int lkmax = qi + kWin;                 // j<=q  (<=79)

    int   lks[3] = { lane, lane + 32, lane + 64 };
    float sc[3];
    float mloc = -3.0e38f;
#pragma unroll
    for (int t = 0; t < 3; ++t) {
      const int lk = lks[t];
      float sval = -3.0e38f;
      if (lk < 80 && lk >= lkmin && lk <= lkmax) {
        float acc = 0.f;
        for (int d = 0; d < kDk; d += 4) {
          const float4 qv = *(const float4*)&sQ[qi][d];
          const v4bf   kk = *(const v4bf*)&sK[lk][d];
          acc += qv.x * (float)kk.x + qv.y * (float)kk.y
               + qv.z * (float)kk.z + qv.w * (float)kk.w;
        }
        sval = acc * scale;
      }
      sc[t] = sval;
      mloc  = fmaxf(mloc, sval);
    }
#pragma unroll
    for (int o = 16; o; o >>= 1) mloc = fmaxf(mloc, __shfl_xor(mloc, o));

    float ssum = 0.f;
    float p[3];
#pragma unroll
    for (int t = 0; t < 3; ++t) {
      p[t] = (sc[t] > -1.0e37f) ? __expf(sc[t] - mloc) : 0.f;
      ssum += p[t];
    }
#pragma unroll
    for (int o = 16; o; o >>= 1) ssum += __shfl_xor(ssum, o);

#pragma unroll
    for (int t = 0; t < 3; ++t)
      if (lks[t] < 80) sP[qi][lks[t]] = p[t];
    __syncthreads();

    const int d = lane * 4;
    float o0 = 0.f, o1 = 0.f, o2 = 0.f, o3 = 0.f;
    for (int lk = 0; lk < 80; ++lk) {
      const float pp = sP[qi][lk];
      const v4bf  vv = *(const v4bf*)&sV[lk][d];
      o0 += pp * (float)vv.x;
      o1 += pp * (float)vv.y;
      o2 += pp * (float)vv.z;
      o3 += pp * (float)vv.w;
    }
    const float r = 1.0f / ssum;
    v4bf outv;
    outv.x = (bf16)(o0 * r); outv.y = (bf16)(o1 * r);
    outv.z = (bf16)(o2 * r); outv.w = (bf16)(o3 * r);
    *(v4bf*)(ob + (((size_t)(b * kS + qglob)) * kH + h) * kDk + d) = outv;
    __syncthreads();
  }
}

// ---------------- launch ----------------
extern "C" void kernel_launch(void* const* d_in, const int* in_sizes, int n_in,
                              void* d_out, int out_size, void* d_ws, size_t ws_size,
                              hipStream_t stream) {
  (void)in_sizes; (void)n_in; (void)out_size; (void)ws_size;
  const float* x   = (const float*)d_in[0];
  const float* Wq  = (const float*)d_in[1];
  const float* bq  = (const float*)d_in[2];
  const float* Wk  = (const float*)d_in[3];
  const float* bk  = (const float*)d_in[4];
  const float* Wv  = (const float*)d_in[5];
  const float* bv  = (const float*)d_in[6];
  const float* Wo  = (const float*)d_in[7];
  const float* bo  = (const float*)d_in[8];
  const float* qn  = (const float*)d_in[9];
  const float* kn  = (const float*)d_in[10];
  const float* sc  = (const float*)d_in[11];

  uint8_t* ws = (uint8_t*)d_ws;
  size_t off = 0;
  auto carve = [&](size_t bytes) -> void* {
    void* p = ws + off;
    off = (off + bytes + 255) & ~(size_t)255;
    return p;
  };
  // ws layout (~122 MB total)
  bf16* xb   = (bf16*)carve((size_t)kRows * kD  * sizeof(bf16));   // 16 MB
  bf16* Wqb  = (bf16*)carve((size_t)kD    * kD  * sizeof(bf16));   // 8 MB
  bf16* Wkb  = (bf16*)carve((size_t)kKVD  * kD  * sizeof(bf16));   // 2 MB
  bf16* Wvb  = (bf16*)carve((size_t)kKVD  * kD  * sizeof(bf16));   // 2 MB
  bf16* Wob  = (bf16*)carve((size_t)kD    * kD  * sizeof(bf16));   // 8 MB
  float* qraw = (float*)carve((size_t)kRows * kD  * sizeof(float)); // 32 MB
  float* kraw = (float*)carve((size_t)kRows * kKVD * sizeof(float)); // 8 MB
  bf16* vb   = (bf16*)carve((size_t)kRows * kKVD * sizeof(bf16));  // 4 MB
  bf16* qb   = (bf16*)carve((size_t)kRows * kD  * sizeof(bf16));   // 16 MB
  bf16* kb   = (bf16*)carve((size_t)kRows * kKVD * sizeof(bf16));  // 4 MB
  bf16* ob   = (bf16*)carve((size_t)kRows * kD  * sizeof(bf16));   // 16 MB

  // 1) bf16 casts
  {
    int n4;
    n4 = kRows * kD / 4;  cvt_f32_to_bf16<<<(n4 + 255) / 256, 256, 0, stream>>>(x,  xb,  n4);
    n4 = kD * kD / 4;     cvt_f32_to_bf16<<<(n4 + 255) / 256, 256, 0, stream>>>(Wq, Wqb, n4);
    n4 = kKVD * kD / 4;   cvt_f32_to_bf16<<<(n4 + 255) / 256, 256, 0, stream>>>(Wk, Wkb, n4);
    n4 = kKVD * kD / 4;   cvt_f32_to_bf16<<<(n4 + 255) / 256, 256, 0, stream>>>(Wv, Wvb, n4);
    n4 = kD * kD / 4;     cvt_f32_to_bf16<<<(n4 + 255) / 256, 256, 0, stream>>>(Wo, Wob, n4);
  }

  // 2) QKV projections (WMMA + TDM staging)
  gemm_bf16_wmma<<<dim3(kRows / 128, kD / 128),   256, 0, stream>>>(xb, Wqb, bq, qraw, kRows, kD,   kD, 0);
  gemm_bf16_wmma<<<dim3(kRows / 128, kKVD / 128), 256, 0, stream>>>(xb, Wkb, bk, kraw, kRows, kKVD, kD, 0);
  gemm_bf16_wmma<<<dim3(kRows / 128, kKVD / 128), 256, 0, stream>>>(xb, Wvb, bv, vb,   kRows, kKVD, kD, 1);

  // 3) RMSNorm + RoPE -> bf16 q/k
  {
    int totq = kRows * kH;   // 65536 vectors, 8 waves/block
    int totk = kRows * kKVH; // 16384
    normrope_kernel<<<totq / 8, 256, 0, stream>>>(qraw, qn, qb, kH,   totq);
    normrope_kernel<<<totk / 8, 256, 0, stream>>>(kraw, kn, kb, kKVH, totk);
  }

  // 4) windowed causal attention
  attn_kernel<<<kB * kH * (kS / 16), 256, 0, stream>>>(qb, kb, vb, sc, ob);

  // 5) output projection (WMMA + TDM staging) -> d_out (f32)
  gemm_bf16_wmma<<<dim3(kRows / 128, kD / 128), 256, 0, stream>>>(ob, Wob, bo, (float*)d_out,
                                                                  kRows, kD, kD, 0);
}
